// TC_Encoder_3942779978188
// MI455X (gfx1250) — matrix operand order repaired and checked by reading, not compiled, not run on hardware
//
#include <hip/hip_runtime.h>
#include <hip/hip_bf16.h>

// ---------------------------------------------------------------------------
// MI455X (gfx1250) implementation.
//   feat = concat(branch1, branch2)  [512 x 92160]  (fp32, in workspace)
//   y    = relu(bn(feat @ W + b))    [512 x 512]
// FC GEMM staging:
//   A tile (128x32 f32, strided)  -> TENSOR_LOAD_TO_LDS (TDM, TENSORcnt), 1 instr
//   B tile (32x64 f32, contiguous)-> GLOBAL_LOAD_ASYNC_TO_LDS_B128 (ASYNCcnt)
// Compute: f32->bf16 hi/lo split at fragment load, bf16x3 WMMA, f32 accum.
// ---------------------------------------------------------------------------

typedef __attribute__((ext_vector_type(16))) __bf16 v16bf;
typedef __attribute__((ext_vector_type(8)))  __bf16 v8bf;
typedef __attribute__((ext_vector_type(8)))  float  v8f;
typedef __attribute__((ext_vector_type(4)))  float  v4f;
typedef __attribute__((ext_vector_type(4)))  unsigned int u32x4;
typedef __attribute__((ext_vector_type(8)))  unsigned int u32x8;

#define EPSBN 1e-5f
#define Bsz 512
#define Csz 64
#define Tsz 20
#define Kfc 92160           // 64*64*20 + 20*8*64
#define Nfc 512
#define B2OFF 81920

// ======================= branch1: per-channel encoders ======================
// grid (64 channels, 128 batch-tiles of 4), 256 threads
__global__ __launch_bounds__(256) void branch1_kernel(
    const float* __restrict__ x,
    const float* __restrict__ w1, const float* __restrict__ b1,
    const float* __restrict__ g1, const float* __restrict__ be1,
    const float* __restrict__ m1, const float* __restrict__ vv1,
    const float* __restrict__ w2, const float* __restrict__ b2,
    const float* __restrict__ g2, const float* __restrict__ be2,
    const float* __restrict__ m2, const float* __restrict__ vv2,
    const float* __restrict__ w3, const float* __restrict__ b3,
    const float* __restrict__ g3, const float* __restrict__ be3,
    const float* __restrict__ m3, const float* __restrict__ vv3,
    const float* __restrict__ w4, const float* __restrict__ b4,
    const float* __restrict__ g4, const float* __restrict__ be4,
    const float* __restrict__ m4, const float* __restrict__ vv4,
    float* __restrict__ feat)
{
    const int c   = blockIdx.x;
    const int b0  = blockIdx.y * 4;
    const int tid = threadIdx.x;

    __shared__ float xs[4][Tsz + 2];
    __shared__ float h1[4][32][Tsz + 2];
    __shared__ float h2[4][32][Tsz];
    __shared__ float h3[4][64][Tsz];
    __shared__ float w1s[32][3];
    __shared__ float w2s[32][32][3];
    __shared__ float s1[32], t1[32], s2[32], t2[32];
    __shared__ float s3[64], t3[64], s4[64], t4[64];

    for (int i = tid; i < 32 * 3; i += 256)
        w1s[i / 3][i % 3] = w1[(c * 32 + i / 3) * 3 + i % 3];
    for (int i = tid; i < 32 * 32 * 3; i += 256) {
        int o = i / 96, r = i % 96;
        w2s[o][r / 3][r % 3] = w2[((c * 32 + o) * 32 + r / 3) * 3 + r % 3];
    }
    if (tid < 32) {
        int o = tid, idx = c * 32 + o;
        float s = g1[idx] * rsqrtf(vv1[idx] + EPSBN);
        s1[o] = s; t1[o] = be1[idx] + s * (b1[idx] - m1[idx]);
        s = g2[idx] * rsqrtf(vv2[idx] + EPSBN);
        s2[o] = s; t2[o] = be2[idx] + s * (b2[idx] - m2[idx]);
    }
    if (tid < 64) {
        int o = tid, idx = c * 64 + o;
        float s = g3[idx] * rsqrtf(vv3[idx] + EPSBN);
        s3[o] = s; t3[o] = be3[idx] + s * (b3[idx] - m3[idx]);
        s = g4[idx] * rsqrtf(vv4[idx] + EPSBN);
        s4[o] = s; t4[o] = be4[idx] + s * (b4[idx] - m4[idx]);
    }
    for (int i = tid; i < 4 * (Tsz + 2); i += 256) {
        int bl = i / (Tsz + 2), tt = i % (Tsz + 2);
        float val = 0.f;
        if (tt >= 1 && tt <= Tsz)
            val = x[((size_t)(b0 + bl) * Csz + c) * Tsz + (tt - 1)];
        xs[bl][tt] = val;
    }
    __syncthreads();

    for (int i = tid; i < 4 * 32 * Tsz; i += 256) {
        int bl = i / (32 * Tsz), r = i % (32 * Tsz), o = r / Tsz, tt = r % Tsz;
        float a = w1s[o][0] * xs[bl][tt] + w1s[o][1] * xs[bl][tt + 1] +
                  w1s[o][2] * xs[bl][tt + 2];
        h1[bl][o][tt + 1] = fmaxf(s1[o] * a + t1[o], 0.f);
    }
    for (int i = tid; i < 4 * 32; i += 256) {
        h1[i / 32][i % 32][0] = 0.f;
        h1[i / 32][i % 32][Tsz + 1] = 0.f;
    }
    __syncthreads();

    for (int i = tid; i < 4 * 32 * Tsz; i += 256) {
        int bl = i / (32 * Tsz), r = i % (32 * Tsz), o = r / Tsz, tt = r % Tsz;
        float a = 0.f;
#pragma unroll
        for (int ci = 0; ci < 32; ++ci)
            a += w2s[o][ci][0] * h1[bl][ci][tt] +
                 w2s[o][ci][1] * h1[bl][ci][tt + 1] +
                 w2s[o][ci][2] * h1[bl][ci][tt + 2];
        h2[bl][o][tt] = fmaxf(s2[o] * a + t2[o], 0.f);
    }
    __syncthreads();

    for (int i = tid; i < 4 * 64 * Tsz; i += 256) {
        int bl = i / (64 * Tsz), r = i % (64 * Tsz), o = r / Tsz, tt = r % Tsz;
        const float* wr = &w3[(c * 64 + o) * 32];
        float a = 0.f;
#pragma unroll
        for (int ci = 0; ci < 32; ++ci) a += wr[ci] * h2[bl][ci][tt];
        h3[bl][o][tt] = fmaxf(s3[o] * a + t3[o], 0.f);
    }
    __syncthreads();

    for (int i = tid; i < 4 * 64 * Tsz; i += 256) {
        int bl = i / (64 * Tsz), r = i % (64 * Tsz), o = r / Tsz, tt = r % Tsz;
        const float* wr = &w4[(c * 64 + o) * 64];
        float a = 0.f;
#pragma unroll
        for (int ci = 0; ci < 64; ++ci) a += wr[ci] * h3[bl][ci][tt];
        feat[(size_t)(b0 + bl) * Kfc + (c * 64 + o) * Tsz + tt] =
            s4[o] * a + t4[o];
    }
}

// ====================== branch2: per-timestep encoders ======================
// grid (20 timesteps, 64 batch-tiles of 8), 256 threads
__global__ __launch_bounds__(256) void branch2_kernel(
    const float* __restrict__ x,
    const float* __restrict__ w1, const float* __restrict__ b1,
    const float* __restrict__ g1, const float* __restrict__ be1,
    const float* __restrict__ m1, const float* __restrict__ vv1,
    const float* __restrict__ w2, const float* __restrict__ b2,
    const float* __restrict__ g2, const float* __restrict__ be2,
    const float* __restrict__ m2, const float* __restrict__ vv2,
    const float* __restrict__ w3, const float* __restrict__ b3,
    const float* __restrict__ g3, const float* __restrict__ be3,
    const float* __restrict__ m3, const float* __restrict__ vv3,
    const float* __restrict__ w4, const float* __restrict__ b4,
    const float* __restrict__ g4, const float* __restrict__ be4,
    const float* __restrict__ m4, const float* __restrict__ vv4,
    float* __restrict__ feat)
{
    const int t   = blockIdx.x;
    const int b0  = blockIdx.y * 8;
    const int tid = threadIdx.x;

    __shared__ float xs[8][Csz + 2];
    __shared__ float h1[8][4][Csz + 2];
    __shared__ float h2[8][4][Csz];
    __shared__ float h3[8][8][Csz];
    __shared__ float w1s[4][3], w2s[4][4][3], w3s[8][4], w4s[8][8];
    __shared__ float s1[4], t1c[4], s2[4], t2c[4], s3[8], t3c[8], s4[8], t4c[8];

    if (tid < 12)  w1s[tid / 3][tid % 3] = w1[(t * 4 + tid / 3) * 3 + tid % 3];
    if (tid < 48) { int o = tid / 12, r = tid % 12;
        w2s[o][r / 3][r % 3] = w2[((t * 4 + o) * 4 + r / 3) * 3 + r % 3]; }
    if (tid < 32)  w3s[tid / 4][tid % 4] = w3[(t * 8 + tid / 4) * 4 + tid % 4];
    if (tid < 64)  w4s[tid / 8][tid % 8] = w4[(t * 8 + tid / 8) * 8 + tid % 8];
    if (tid < 4) {
        int o = tid, idx = t * 4 + o;
        float s = g1[idx] * rsqrtf(vv1[idx] + EPSBN);
        s1[o] = s; t1c[o] = be1[idx] + s * (b1[idx] - m1[idx]);
        s = g2[idx] * rsqrtf(vv2[idx] + EPSBN);
        s2[o] = s; t2c[o] = be2[idx] + s * (b2[idx] - m2[idx]);
    }
    if (tid < 8) {
        int o = tid, idx = t * 8 + o;
        float s = g3[idx] * rsqrtf(vv3[idx] + EPSBN);
        s3[o] = s; t3c[o] = be3[idx] + s * (b3[idx] - m3[idx]);
        s = g4[idx] * rsqrtf(vv4[idx] + EPSBN);
        s4[o] = s; t4c[o] = be4[idx] + s * (b4[idx] - m4[idx]);
    }
    for (int i = tid; i < 8 * (Csz + 2); i += 256) {
        int bl = i / (Csz + 2), p = i % (Csz + 2);
        float val = 0.f;
        if (p >= 1 && p <= Csz)
            val = x[((size_t)(b0 + bl) * Csz + (p - 1)) * Tsz + t];
        xs[bl][p] = val;
    }
    __syncthreads();

    for (int i = tid; i < 8 * 4 * Csz; i += 256) {
        int bl = i >> 8, r = i & 255, o = r >> 6, p = r & 63;
        float a = w1s[o][0] * xs[bl][p] + w1s[o][1] * xs[bl][p + 1] +
                  w1s[o][2] * xs[bl][p + 2];
        h1[bl][o][p + 1] = fmaxf(s1[o] * a + t1c[o], 0.f);
    }
    for (int i = tid; i < 32; i += 256) {
        h1[i / 4][i % 4][0] = 0.f;
        h1[i / 4][i % 4][Csz + 1] = 0.f;
    }
    __syncthreads();

    for (int i = tid; i < 8 * 4 * Csz; i += 256) {
        int bl = i >> 8, r = i & 255, o = r >> 6, p = r & 63;
        float a = 0.f;
#pragma unroll
        for (int ci = 0; ci < 4; ++ci)
            a += w2s[o][ci][0] * h1[bl][ci][p] +
                 w2s[o][ci][1] * h1[bl][ci][p + 1] +
                 w2s[o][ci][2] * h1[bl][ci][p + 2];
        h2[bl][o][p] = fmaxf(s2[o] * a + t2c[o], 0.f);
    }
    __syncthreads();

    for (int i = tid; i < 8 * 8 * Csz; i += 256) {
        int bl = i >> 9, r = i & 511, o = r >> 6, p = r & 63;
        float a = 0.f;
#pragma unroll
        for (int ci = 0; ci < 4; ++ci) a += w3s[o][ci] * h2[bl][ci][p];
        h3[bl][o][p] = fmaxf(s3[o] * a + t3c[o], 0.f);
    }
    __syncthreads();

    for (int i = tid; i < 8 * 8 * Csz; i += 256) {
        int bl = i >> 9, r = i & 511, o = r >> 6, p = r & 63;
        float a = 0.f;
#pragma unroll
        for (int ci = 0; ci < 8; ++ci) a += w4s[o][ci] * h3[bl][ci][p];
        feat[(size_t)(b0 + bl) * Kfc + B2OFF + (t * 8 + o) * Csz + p] =
            s4[o] * a + t4c[o];
    }
}

// ============================ FC GEMM (WMMA) ================================
#define BM 128
#define BN 64
#define BK 32
#define KSPLIT 16
#define KCHUNK (Kfc / KSPLIT)     // 5760
#define NSTEP  (KCHUNK / BK)      // 180 k-steps

// ---- TDM: one-instruction 2D strided tile DMA, memory -> LDS (TENSORcnt) ---
// Builds D# group0 (4 SGPRs) + group1 (8 SGPRs) per ISA 8.3/8.4; 2D tile, no
// padding/iteration/multicast, data_size = 4 bytes, type = 2 ("image").
__device__ __forceinline__ void tdm_load_2d(
    unsigned lds_off, const void* gptr,
    unsigned tensor_d0, unsigned tensor_d1, unsigned stride0,
    unsigned tile_d0, unsigned tile_d1)
{
    unsigned long long ga = (unsigned long long)gptr;
    u32x4 g0;
    g0[0] = 1u;                                             // count=1 (user D#)
    g0[1] = lds_off;                                        // lds_addr
    g0[2] = (unsigned)(ga & 0xFFFFFFFFu);                   // global_addr lo
    g0[3] = (unsigned)((ga >> 32) & 0x01FFFFFFu) | 0x80000000u; // addr hi | type=2
    u32x8 g1;
    g1[0] = 0x00020000u;                                    // data_size=4B
    g1[1] = (tensor_d0 & 0xFFFFu) << 16;                    // tensor_dim0 lo16
    g1[2] = (tensor_d0 >> 16) | ((tensor_d1 & 0xFFFFu) << 16);
    g1[3] = (tensor_d1 >> 16) | (tile_d0 << 16);            // tile_dim0
    g1[4] = tile_d1 & 0xFFFFu;                              // tile_dim1 (dim2=0)
    g1[5] = stride0;                                        // tensor_dim0_stride lo32
    g1[6] = 0u;
    g1[7] = 0u;
    asm volatile("tensor_load_to_lds %0, %1" :: "s"(g0), "s"(g1) : "memory");
}

// memory -> LDS, 16B per lane (ASYNCcnt)
__device__ __forceinline__ void async_copy16(unsigned lds_off,
                                             const float* gbase,
                                             unsigned gofs_bytes) {
    asm volatile("global_load_async_to_lds_b128 %0, %1, %2"
                 :: "v"(lds_off), "v"(gofs_bytes), "s"(gbase)
                 : "memory");
}

__device__ __forceinline__ void wait_async0() {
    asm volatile("s_wait_asynccnt 0x0" ::: "memory");
}

// f32[16] -> bf16 hi + bf16 residual lo (packed cvt on VALU)
__device__ __forceinline__ void cvt_split16(const float* f, v16bf& hi, v16bf& lo) {
    union VB { v16bf v; __bf16 e[16]; } h, l;
#pragma unroll
    for (int j = 0; j < 16; ++j) {
        float x   = f[j];
        __bf16 hh = (__bf16)x;
        h.e[j] = hh;
        l.e[j] = (__bf16)(x - (float)hh);
    }
    hi = h.v; lo = l.v;
}

// grid (4, 8, 16), 256 threads (8 wave32s, arranged 4(M) x 2(N))
__global__ __launch_bounds__(256) void fc_gemm_tdm_bf16x3(
    const float* __restrict__ A,   // feat [512][92160]
    const float* __restrict__ W,   // [92160][512]
    float* __restrict__ acc)       // [512][512] f32 accumulator (pre-zeroed)
{
    __shared__ __align__(16) float As[2][BM][BK];   // 2 x 16 KB (TDM-staged)
    __shared__ __align__(16) float Bs[2][BK][BN];   // 2 x  8 KB (async-staged)

    const int tid   = threadIdx.x;
    const int m0    = blockIdx.x * BM;
    const int n0    = blockIdx.y * BN;
    const int kbase = blockIdx.z * KCHUNK;

    const int wid   = tid >> 5;
    const int lane  = tid & 31;
    const int wm    = wid >> 1;      // 0..3 -> 32 rows each
    const int wn    = wid & 1;       // 0..1 -> 32 cols each
    const int lrow  = lane & 15;
    const int lhalf = lane >> 4;

    v8f accf[2][2];
#pragma unroll
    for (int i = 0; i < 2; ++i)
#pragma unroll
        for (int j = 0; j < 2; ++j) accf[i][j] = (v8f){};

    // A tile: one TDM descriptor (wave 0 issues; TDM ignores EXEC, so only
    // one wave must issue or the DMA would be duplicated)
    auto stage_A = [&](int buf, int kb) {
        tdm_load_2d((unsigned)(uintptr_t)&As[buf][0][0],
                    A + (size_t)m0 * Kfc + kb,
                    /*tensor_d0=*/Kfc, /*tensor_d1=*/Bsz, /*stride0=*/Kfc,
                    /*tile_d0=*/BK, /*tile_d1=*/BM);
    };
    // B tile: 512 x 16B lane-async copies (2 per thread)
    auto stage_B = [&](int buf, int kb) {
        unsigned bbase = (unsigned)(uintptr_t)&Bs[buf][0][0];
        for (int s = tid; s < (BK * BN) / 4; s += 256) {
            int krow = s >> 4, q = s & 15;
            async_copy16(bbase + (unsigned)(krow * (BN * 4) + q * 16), W,
                         (unsigned)(((size_t)(kb + krow) * Nfc + n0) * 4 + q * 16));
        }
    };

    if (wid == 0) stage_A(0, kbase);
    stage_B(0, kbase);

    for (int ks = 0; ks < NSTEP; ++ks) {
        const int cur = ks & 1;
        wait_async0();                                   // my B copies landed
        if (wid == 0) __builtin_amdgcn_s_wait_tensorcnt(0);  // A DMA landed
        __syncthreads();                                 // publish LDS to all waves
        if (ks + 1 < NSTEP) {                            // overlap next tile's DMA
            if (wid == 0) stage_A(cur ^ 1, kbase + (ks + 1) * BK);
            stage_B(cur ^ 1, kbase + (ks + 1) * BK);
        }

        // ---- A fragments: 16x32 A layout (lanes 0-15: K 0-7/16-23; 16-31: 8-15/24-31)
        v16bf a_hi[2], a_lo[2], b_hi[2], b_lo[2];
#pragma unroll
        for (int i = 0; i < 2; ++i) {
            int row = wm * 32 + i * 16 + lrow;
            int kh  = lhalf * 8;
            float af[16];
#pragma unroll
            for (int q = 0; q < 8; ++q) af[q] = As[cur][row][kh + q];
#pragma unroll
            for (int q = 0; q < 8; ++q) af[8 + q] = As[cur][row][kh + 16 + q];
            cvt_split16(af, a_hi[i], a_lo[i]);
        }
        // ---- B fragments: 32x16 B layout (lanes 0-15: K 0-7/16-23; 16-31: 8-15/24-31)
#pragma unroll
        for (int j = 0; j < 2; ++j) {
            int col  = wn * 32 + j * 16 + lrow;
            int kcol = lhalf * 8;
            float bfv[16];
#pragma unroll
            for (int q = 0; q < 16; ++q) bfv[q] = Bs[cur][kcol + q + (q >> 3) * 8][col];
            cvt_split16(bfv, b_hi[j], b_lo[j]);
        }

        // ---- bf16x3: hi*hi + hi*lo + lo*hi ----
#pragma unroll
        for (int i = 0; i < 2; ++i)
#pragma unroll
            for (int j = 0; j < 2; ++j) {
                accf[i][j] = __builtin_amdgcn_wmma_f32_16x16x32_bf16(
                    false, a_hi[i], false, b_hi[j], (short)0, accf[i][j], false, false);
                accf[i][j] = __builtin_amdgcn_wmma_f32_16x16x32_bf16(
                    false, a_hi[i], false, b_lo[j], (short)0, accf[i][j], false, false);
                accf[i][j] = __builtin_amdgcn_wmma_f32_16x16x32_bf16(
                    false, a_lo[i], false, b_hi[j], (short)0, accf[i][j], false, false);
            }
    }

    // ---- split-K accumulate (C frag: lane n = lrow; VGPR v -> M = v + 8*lhalf)
    union F8 { v8f v; float f[8]; };
#pragma unroll
    for (int i = 0; i < 2; ++i)
#pragma unroll
        for (int j = 0; j < 2; ++j) {
            int n  = n0 + wn * 32 + j * 16 + lrow;
            int mb = m0 + wm * 32 + i * 16 + lhalf * 8;
            F8 c; c.v = accf[i][j];
#pragma unroll
            for (int v = 0; v < 8; ++v)
                atomicAdd(&acc[(size_t)(mb + v) * Nfc + n], c.f[v]);
        }
}

// ============================= epilogue / util ==============================
__global__ __launch_bounds__(256) void zero_kernel(float* p, int n) {
    int i = blockIdx.x * 256 + threadIdx.x;
    if (i < n) p[i] = 0.f;
}

__global__ __launch_bounds__(256) void finalize_kernel(
    const float* __restrict__ acc, const float* __restrict__ bias,
    const float* __restrict__ g, const float* __restrict__ be,
    const float* __restrict__ m, const float* __restrict__ v,
    float* __restrict__ out)
{
    int i = blockIdx.x * 256 + threadIdx.x;   // 0 .. 512*512-1
    int n = i & (Nfc - 1);
    float s = g[n] * rsqrtf(v[n] + EPSBN);
    out[i] = fmaxf((acc[i] + bias[n] - m[n]) * s + be[n], 0.f);
}

// ================================ launcher ==================================
extern "C" void kernel_launch(void* const* d_in, const int* in_sizes, int n_in,
                              void* d_out, int out_size, void* d_ws, size_t ws_size,
                              hipStream_t stream) {
    (void)in_sizes; (void)n_in; (void)out_size; (void)ws_size;
    const float* x = (const float*)d_in[0];
    const float* c_w1 = (const float*)d_in[1];  const float* c_b1 = (const float*)d_in[2];
    const float* c_g1 = (const float*)d_in[3];  const float* c_be1 = (const float*)d_in[4];
    const float* c_m1 = (const float*)d_in[5];  const float* c_v1 = (const float*)d_in[6];
    const float* c_w2 = (const float*)d_in[7];  const float* c_b2 = (const float*)d_in[8];
    const float* c_g2 = (const float*)d_in[9];  const float* c_be2 = (const float*)d_in[10];
    const float* c_m2 = (const float*)d_in[11]; const float* c_v2 = (const float*)d_in[12];
    const float* c_w3 = (const float*)d_in[13]; const float* c_b3 = (const float*)d_in[14];
    const float* c_g3 = (const float*)d_in[15]; const float* c_be3 = (const float*)d_in[16];
    const float* c_m3 = (const float*)d_in[17]; const float* c_v3 = (const float*)d_in[18];
    const float* c_w4 = (const float*)d_in[19]; const float* c_b4 = (const float*)d_in[20];
    const float* c_g4 = (const float*)d_in[21]; const float* c_be4 = (const float*)d_in[22];
    const float* c_m4 = (const float*)d_in[23]; const float* c_v4 = (const float*)d_in[24];
    const float* t_w1 = (const float*)d_in[25]; const float* t_b1 = (const float*)d_in[26];
    const float* t_g1 = (const float*)d_in[27]; const float* t_be1 = (const float*)d_in[28];
    const float* t_m1 = (const float*)d_in[29]; const float* t_v1 = (const float*)d_in[30];
    const float* t_w2 = (const float*)d_in[31]; const float* t_b2 = (const float*)d_in[32];
    const float* t_g2 = (const float*)d_in[33]; const float* t_be2 = (const float*)d_in[34];
    const float* t_m2 = (const float*)d_in[35]; const float* t_v2 = (const float*)d_in[36];
    const float* t_w3 = (const float*)d_in[37]; const float* t_b3 = (const float*)d_in[38];
    const float* t_g3 = (const float*)d_in[39]; const float* t_be3 = (const float*)d_in[40];
    const float* t_m3 = (const float*)d_in[41]; const float* t_v3 = (const float*)d_in[42];
    const float* t_w4 = (const float*)d_in[43]; const float* t_b4 = (const float*)d_in[44];
    const float* t_g4 = (const float*)d_in[45]; const float* t_be4 = (const float*)d_in[46];
    const float* t_m4 = (const float*)d_in[47]; const float* t_v4 = (const float*)d_in[48];
    const float* fc_W = (const float*)d_in[49]; const float* fc_b = (const float*)d_in[50];
    const float* fc_g = (const float*)d_in[51]; const float* fc_be = (const float*)d_in[52];
    const float* fc_m = (const float*)d_in[53]; const float* fc_v = (const float*)d_in[54];

    float* feat = (float*)d_ws;                                   // 512*92160*4 B
    float* acc  = (float*)((char*)d_ws + (size_t)Bsz * Kfc * sizeof(float)); // 1 MB

    branch1_kernel<<<dim3(Csz, Bsz / 4), 256, 0, stream>>>(
        x, c_w1, c_b1, c_g1, c_be1, c_m1, c_v1,
           c_w2, c_b2, c_g2, c_be2, c_m2, c_v2,
           c_w3, c_b3, c_g3, c_be3, c_m3, c_v3,
           c_w4, c_b4, c_g4, c_be4, c_m4, c_v4, feat);

    branch2_kernel<<<dim3(Tsz, Bsz / 8), 256, 0, stream>>>(
        x, t_w1, t_b1, t_g1, t_be1, t_m1, t_v1,
           t_w2, t_b2, t_g2, t_be2, t_m2, t_v2,
           t_w3, t_b3, t_g3, t_be3, t_m3, t_v3,
           t_w4, t_b4, t_g4, t_be4, t_m4, t_v4, feat);

    zero_kernel<<<(Bsz * Nfc) / 256, 256, 0, stream>>>(acc, Bsz * Nfc);

    fc_gemm_tdm_bf16x3<<<dim3(Bsz / BM, Nfc / BN, KSPLIT), 256, 0, stream>>>(
        feat, fc_W, acc);

    finalize_kernel<<<(Bsz * Nfc) / 256, 256, 0, stream>>>(
        acc, fc_b, fc_g, fc_be, fc_m, fc_v, (float*)d_out);
}
